// MSVABiLstm_17377437680005
// MI455X (gfx1250) — matrix operand order repaired and checked by reading, not compiled, not run on hardware
//
#include <hip/hip_runtime.h>
#include <hip/hip_bf16.h>
#include <math.h>

#define SEQ   4096
#define DM    1024
#define HD    1024
#define G4    (4 * HD)
#define APER  250
#define BT    16          // tile radius covering the band (|ti-tj| <= 16)
#define LN_EPS 1e-6f

typedef _Float16 f16;
typedef __attribute__((ext_vector_type(16))) _Float16 v16h;
typedef __attribute__((ext_vector_type(8)))  float    v8f;

// ---------------------------------------------------------------------------
// WMMA fragment helpers (CDNA5 16x16x32 f16 layout, wave32)
// A 16x32 fp16 frag: lanes 0-15 -> rows, K {0..7,16..23}; lanes 16-31 -> K {8..15,24..31}
// ---------------------------------------------------------------------------
__device__ inline v16h frag_rm(const float* __restrict__ P, int r0, int k0, int ld) {
    int lane = threadIdx.x & 31;
    const float* p = P + (size_t)(r0 + (lane & 15)) * ld + k0 + ((lane >> 4) << 3);
    const float4* q = (const float4*)p;
    float4 a0 = q[0], a1 = q[1], a2 = q[4], a3 = q[5];
    v16h f;
    f[0]=(f16)a0.x; f[1]=(f16)a0.y; f[2]=(f16)a0.z; f[3]=(f16)a0.w;
    f[4]=(f16)a1.x; f[5]=(f16)a1.y; f[6]=(f16)a1.z; f[7]=(f16)a1.w;
    f[8]=(f16)a2.x; f[9]=(f16)a2.y; f[10]=(f16)a2.z; f[11]=(f16)a2.w;
    f[12]=(f16)a3.x; f[13]=(f16)a3.y; f[14]=(f16)a3.z; f[15]=(f16)a3.w;
    return f;
}

// logical A[r][k] = P[k*ld + r]  (used for att^T as the A operand)
__device__ inline v16h frag_cm(const float* __restrict__ P, int r0, int k0, int ld) {
    int lane = threadIdx.x & 31;
    int r  = r0 + (lane & 15);
    int kb = k0 + ((lane >> 4) << 3);
    v16h f;
#pragma unroll
    for (int i = 0; i < 8; ++i) f[i]     = (f16)P[(size_t)(kb + i) * ld + r];
#pragma unroll
    for (int i = 0; i < 8; ++i) f[8 + i] = (f16)P[(size_t)(kb + 16 + i) * ld + r];
    return f;
}

__device__ inline v8f wmma_f16(v16h a, v16h b, v8f c) {
    return __builtin_amdgcn_wmma_f32_16x16x32_f16(false, a, false, b, (short)0, c,
                                                  false, false);
}

// ---------------------------------------------------------------------------
// Generic C[M,N] = A[M,K] @ W[N,K]^T with fused epilogue
// ---------------------------------------------------------------------------
enum { EP_SCALE = 1, EP_BIAS = 2, EP_BIAS2 = 4, EP_RELU = 8 };

template <int EP>
__global__ __launch_bounds__(128)
void wmma_gemm_nt(const float* __restrict__ A, const float* __restrict__ W,
                  float* __restrict__ C, int Mr, int Nc, int Kd,
                  float alpha, const float* __restrict__ b1,
                  const float* __restrict__ b2) {
    int wid = blockIdx.x * 4 + (threadIdx.x >> 5);
    int nw  = gridDim.x * 4;
    int tN  = Nc >> 4;
    int ntiles = (Mr >> 4) * tN;
    int lane = threadIdx.x & 31;
    for (int t = wid; t < ntiles; t += nw) {
        int m0 = (t / tN) << 4, n0 = (t % tN) << 4;
        v8f acc = {};
        for (int k0 = 0; k0 < Kd; k0 += 32)
            acc = wmma_f16(frag_rm(A, m0, k0, Kd), frag_rm(W, n0, k0, Kd), acc);
        int n  = n0 + (lane & 15);
        int mb = m0 + ((lane >> 4) << 3);
        float bias = 0.f;
        if (EP & EP_BIAS)  bias += b1[n];
        if (EP & EP_BIAS2) bias += b2[n];
#pragma unroll
        for (int r = 0; r < 8; ++r) {
            float v = acc[r];
            if (EP & EP_SCALE) v *= alpha;
            v += bias;
            if (EP & EP_RELU) v = fmaxf(v, 0.f);
            C[(size_t)(mb + r) * Nc + n] = v;
        }
    }
}

// Banded logits: att[i,j] = (Q K^T)[i,j] only where |i-j| < APER (rest stays 0)
__global__ __launch_bounds__(128)
void wmma_logits_band(const float* __restrict__ Q, const float* __restrict__ K,
                      float* __restrict__ att) {
    const int nd = 2 * BT + 1;                 // 33 candidate tile-cols per tile-row
    int wid = blockIdx.x * 4 + (threadIdx.x >> 5);
    if (wid >= (SEQ / 16) * nd) return;        // wave-uniform: EXEC stays full
    int ti = wid / nd;
    int tj = ti - BT + (wid % nd);
    if (tj < 0 || tj >= SEQ / 16) return;      // wave-uniform
    int m0 = ti << 4, n0 = tj << 4;
    v8f acc = {};
    for (int k0 = 0; k0 < DM; k0 += 32)
        acc = wmma_f16(frag_rm(Q, m0, k0, DM), frag_rm(K, n0, k0, DM), acc);
    int lane = threadIdx.x & 31;
    int n  = n0 + (lane & 15);
    int mb = m0 + ((lane >> 4) << 3);
#pragma unroll
    for (int r = 0; r < 8; ++r) {
        int m = mb + r;
        int d = m > n ? m - n : n - m;
        if (d < APER) att[(size_t)m * SEQ + n] = acc[r];
    }
}

// C[SEQ, DM] = att^T @ V, K-loop restricted to the band around each row tile
__global__ __launch_bounds__(128)
void wmma_attT_v(const float* __restrict__ att, const float* __restrict__ V,
                 float* __restrict__ C) {
    int wid = blockIdx.x * 4 + (threadIdx.x >> 5);
    int nw  = gridDim.x * 4;
    const int tN = DM >> 4;
    int lane = threadIdx.x & 31;
    for (int t = wid; t < (SEQ >> 4) * tN; t += nw) {
        int m0 = (t / tN) << 4, n0 = (t % tN) << 4;
        int klo = m0 - (APER - 1); if (klo < 0) klo = 0; klo &= ~31;
        int khi = m0 + 15 + APER;  if (khi > SEQ) khi = SEQ;
        khi = (khi + 31) & ~31;    if (khi > SEQ) khi = SEQ;
        v8f acc = {};
        for (int k0 = klo; k0 < khi; k0 += 32)
            acc = wmma_f16(frag_cm(att, m0, k0, SEQ), frag_cm(V, n0, k0, DM), acc);
        int n  = n0 + (lane & 15);
        int mb = m0 + ((lane >> 4) << 3);
#pragma unroll
        for (int r = 0; r < 8; ++r) C[(size_t)(mb + r) * DM + n] = acc[r];
    }
}

// ---------------------------------------------------------------------------
// Block reductions (blockDim == 256)
// ---------------------------------------------------------------------------
__device__ inline float blk_sum(float v, float* red) {
    int tid = threadIdx.x;
    red[tid] = v; __syncthreads();
    for (int s = 128; s; s >>= 1) { if (tid < s) red[tid] += red[tid + s]; __syncthreads(); }
    float r = red[0]; __syncthreads();
    return r;
}
__device__ inline float blk_max(float v, float* red) {
    int tid = threadIdx.x;
    red[tid] = v; __syncthreads();
    for (int s = 128; s; s >>= 1) { if (tid < s) red[tid] = fmaxf(red[tid], red[tid + s]); __syncthreads(); }
    float r = red[0]; __syncthreads();
    return r;
}

// Row-wise softmax over the in-band columns only; out-of-band stays 0
__global__ __launch_bounds__(256)
void band_softmax(float* __restrict__ att) {
    __shared__ float red[256];
    int i = blockIdx.x, tid = threadIdx.x;
    int lo = i - (APER - 1); if (lo < 0) lo = 0;
    int hi = i + APER;       if (hi > SEQ) hi = SEQ;
    float* row = att + (size_t)i * SEQ;
    float m = -1e30f;
    for (int j = lo + tid; j < hi; j += 256) m = fmaxf(m, row[j]);
    m = blk_max(m, red);
    float s = 0.f;
    for (int j = lo + tid; j < hi; j += 256) { float e = expf(row[j] - m); row[j] = e; s += e; }
    float inv = 1.f / blk_sum(s, red);
    for (int j = lo + tid; j < hi; j += 256) row[j] *= inv;
}

// LayerNorm(Y + X) with ddof=1 std, eps on std; write or accumulate
template <bool ACC>
__global__ __launch_bounds__(256)
void ln_residual(const float* __restrict__ Y, const float* __restrict__ X,
                 const float* __restrict__ g, const float* __restrict__ b,
                 float* __restrict__ out) {
    __shared__ float red[256];
    int row = blockIdx.x, tid = threadIdx.x;
    const float* y = Y + (size_t)row * DM;
    const float* x = X + (size_t)row * DM;
    float v[4];
#pragma unroll
    for (int j = 0; j < 4; ++j) v[j] = y[tid + j * 256] + x[tid + j * 256];
    float mean = blk_sum(v[0] + v[1] + v[2] + v[3], red) * (1.f / DM);
    float q = 0.f;
#pragma unroll
    for (int j = 0; j < 4; ++j) { float d = v[j] - mean; q += d * d; }
    float stdv = sqrtf(blk_sum(q, red) / (float)(DM - 1));
    float inv = 1.f / (stdv + LN_EPS);
#pragma unroll
    for (int j = 0; j < 4; ++j) {
        int idx = tid + j * 256;
        float r = g[idx] * (v[j] - mean) * inv + b[idx];
        if (ACC) out[(size_t)row * DM + idx] += r;
        else     out[(size_t)row * DM + idx]  = r;
    }
}

// Final head: LN(gamma_ka,beta_ka) -> dot(Wkd)+bkd -> sigmoid -> d_out[row]
__global__ __launch_bounds__(256)
void head_kernel(const float* __restrict__ O, const float* __restrict__ g,
                 const float* __restrict__ b, const float* __restrict__ wkd,
                 const float* __restrict__ bkd, float* __restrict__ out) {
    __shared__ float red[256];
    int row = blockIdx.x, tid = threadIdx.x;
    const float* o = O + (size_t)row * HD;
    float v[4];
#pragma unroll
    for (int j = 0; j < 4; ++j) v[j] = o[tid + j * 256];
    float mean = blk_sum(v[0] + v[1] + v[2] + v[3], red) * (1.f / HD);
    float q = 0.f;
#pragma unroll
    for (int j = 0; j < 4; ++j) { float d = v[j] - mean; q += d * d; }
    float stdv = sqrtf(blk_sum(q, red) / (float)(HD - 1));
    float inv = 1.f / (stdv + LN_EPS);
    float dot = 0.f;
#pragma unroll
    for (int j = 0; j < 4; ++j) {
        int idx = tid + j * 256;
        float r = g[idx] * (v[j] - mean) * inv + b[idx];
        dot += r * wkd[idx];
    }
    dot = blk_sum(dot, red);
    if (tid == 0) out[row] = 1.f / (1.f + expf(-(dot + bkd[0])));
}

__global__ void zero_kernel(float* __restrict__ p, size_t n) {
    size_t i  = (size_t)blockIdx.x * blockDim.x + threadIdx.x;
    size_t st = (size_t)gridDim.x * blockDim.x;
    for (; i < n; i += st) p[i] = 0.f;
}

// ---------------------------------------------------------------------------
// Persistent LSTM recurrence. 64 blocks x 256 threads; block b owns 16 hidden
// units. Whh slice (64 rows, 256 KB) is cached in LDS with conflict-free
// padding; cell state lives in registers; h is double-buffered in global with
// one device-scope barrier per step.
// ---------------------------------------------------------------------------
#define LSTM_NBLK 64
#define WL_STRIDE 258          // 256-float segment padded to even, bank-rotating
#define HS_STRIDE 260
constexpr size_t LSTM_LDS_BYTES =
    (size_t)(256 * WL_STRIDE + 4 * HS_STRIDE + 256 + 64) * sizeof(float);

__global__ __launch_bounds__(256)
void lstm_recurrence(const float* __restrict__ Gx, const float* __restrict__ Whh,
                     float* __restrict__ Hout, float* __restrict__ hbuf,
                     int* __restrict__ bar, int T) {
    extern __shared__ float smem[];
    float* wl   = smem;                       // 256 * WL_STRIDE
    float* hs   = wl + 256 * WL_STRIDE;       // 4 * HS_STRIDE
    float* part = hs + 4 * HS_STRIDE;         // 256
    float* gred = part + 256;                 // 64

    const int tid  = threadIdx.x;
    const int r    = tid >> 2;                // gate-row 0..63
    const int p    = tid & 3;                 // K-quarter
    const int ul   = r & 15;                  // unit within block
    const int gate = r >> 4;                  // 0:i 1:f 2:g 3:o
    const int u0   = blockIdx.x * 16;
    const int wrow = gate * HD + u0 + ul;

    {   // cache this thread's 256-wide Whh slice in LDS
        const float* wsrc = Whh + (size_t)wrow * HD + p * 256;
        float* wdst = wl + tid * WL_STRIDE;
        for (int k = 0; k < 256; ++k) wdst[k] = wsrc[k];
    }
    __syncthreads();

    const float2* wv = (const float2*)(wl + tid * WL_STRIDE);
    const float2* hv = (const float2*)(hs + p * HS_STRIDE);
    float c = 0.f;                            // cell state (valid for tid < 16)

    for (int t = 0; t < T; ++t) {
        const float* hprev = hbuf + (((t & 1) ^ 1) * HD);
#pragma unroll
        for (int j = 0; j < 4; ++j) {
            int idx = tid + j * 256;
            hs[(idx >> 8) * HS_STRIDE + (idx & 255)] = hprev[idx];
        }
        __syncthreads();

        float s = 0.f;
#pragma unroll 8
        for (int k = 0; k < 128; ++k) {
            float2 a = wv[k], h2 = hv[k];
            s = fmaf(a.x, h2.x, s);
            s = fmaf(a.y, h2.y, s);
        }
        part[tid] = s;
        __syncthreads();
        if (p == 0)
            gred[r] = part[tid] + part[tid + 1] + part[tid + 2] + part[tid + 3]
                    + Gx[(size_t)t * G4 + wrow];
        __syncthreads();

        if (tid < 16) {
            float gi = gred[tid],      gf = gred[16 + tid];
            float gg = gred[32 + tid], go = gred[48 + tid];
            gi = 1.f / (1.f + expf(-gi));
            gf = 1.f / (1.f + expf(-gf));
            gg = tanhf(gg);
            go = 1.f / (1.f + expf(-go));
            c = gf * c + gi * gg;
            float h = go * tanhf(c);
            hbuf[(t & 1) * HD + u0 + tid] = h;
            Hout[(size_t)t * HD + u0 + tid] = h;
        }
        // device-scope barrier: everyone consumed h_{t-1} (into LDS) and h_t
        // is globally visible before any block starts step t+1
        __syncthreads();
        if (tid == 0) {
            __threadfence();
            atomicAdd(bar, 1);
            int target = (int)gridDim.x * (t + 1);
            while (__hip_atomic_load(bar, __ATOMIC_ACQUIRE,
                                     __HIP_MEMORY_SCOPE_AGENT) < target)
                __builtin_amdgcn_s_sleep(1);
        }
        __syncthreads();
    }
}

// ---------------------------------------------------------------------------
extern "C" void kernel_launch(void* const* d_in, const int* in_sizes, int n_in,
                              void* d_out, int out_size, void* d_ws, size_t ws_size,
                              hipStream_t stream) {
    (void)in_sizes; (void)n_in; (void)out_size;
    const float* x1    = (const float*)d_in[0];
    const float* x2    = (const float*)d_in[1];
    const float* x3    = (const float*)d_in[2];
    const float* Wq    = (const float*)d_in[3];
    const float* Wk    = (const float*)d_in[4];
    const float* Wv    = (const float*)d_in[5];
    const float* Wo    = (const float*)d_in[6];
    const float* gy    = (const float*)d_in[7];
    const float* by    = (const float*)d_in[8];
    const float* Wih0  = (const float*)d_in[9];
    const float* Whh0  = (const float*)d_in[10];
    const float* bih0  = (const float*)d_in[11];
    const float* bhh0  = (const float*)d_in[12];
    const float* Wih1  = (const float*)d_in[13];
    const float* Whh1  = (const float*)d_in[14];
    const float* bih1  = (const float*)d_in[15];
    const float* bhh1  = (const float*)d_in[16];
    const float* Wka   = (const float*)d_in[17];
    const float* bka   = (const float*)d_in[18];
    const float* gka   = (const float*)d_in[19];
    const float* bka2  = (const float*)d_in[20];
    const float* Wkd   = (const float*)d_in[21];
    const float* bkd   = (const float*)d_in[22];

    const size_t MLN = (size_t)SEQ * DM;          // 4M floats
    float* F    = (float*)d_ws;
    float* Kb   = F;                               // projections
    float* Qb   = F + 1 * MLN;
    float* Vb   = F + 2 * MLN;
    float* Cs   = F + 3 * MLN;                     // att^T @ V
    float* Yo   = F + 4 * MLN;                     // Cs @ Wo^T
    float* Ysum = F + 5 * MLN;                     // stream sum after LN
    float* BIG  = F + 6 * MLN;                     // 16M floats: att (s=1,2) then Gx
    float* h0   = Kb;                              // reuse after attention phase
    float* h1   = Qb;
    float* Ob   = Vb;
    float* STATE = F + 10 * MLN;                   // h double buffer + barrier
    float* hbuf  = STATE;
    int*   bar   = (int*)(STATE + 2 * HD);

    if (ws_size < (10 * MLN + 2 * HD + 256) * sizeof(float)) return;

    float* out_o   = (float*)d_out;                // [4096]
    float* att_out = (float*)d_out + SEQ;          // [4096 x 4096]

    const int GB = 2048;                           // GEMM blocks (4 waves each)
    zero_kernel<<<4096, 256, 0, stream>>>(att_out, (size_t)SEQ * SEQ);
    zero_kernel<<<4096, 256, 0, stream>>>(BIG, (size_t)SEQ * SEQ);

    const float* xs[3] = {x1, x2, x3};
    for (int s = 0; s < 3; ++s) {
        const float* X = xs[s];
        float* att = (s == 0) ? att_out : BIG;
        wmma_gemm_nt<0><<<GB, 128, 0, stream>>>(X, Wk, Kb, SEQ, DM, DM, 1.f, nullptr, nullptr);
        wmma_gemm_nt<EP_SCALE><<<GB, 128, 0, stream>>>(X, Wq, Qb, SEQ, DM, DM, 0.06f, nullptr, nullptr);
        wmma_gemm_nt<0><<<GB, 128, 0, stream>>>(X, Wv, Vb, SEQ, DM, DM, 1.f, nullptr, nullptr);
        wmma_logits_band<<<((SEQ / 16) * (2 * BT + 1) + 3) / 4, 128, 0, stream>>>(Qb, Kb, att);
        band_softmax<<<SEQ, 256, 0, stream>>>(att);
        wmma_attT_v<<<GB, 128, 0, stream>>>(att, Vb, Cs);
        wmma_gemm_nt<0><<<GB, 128, 0, stream>>>(Cs, Wo, Yo, SEQ, DM, DM, 1.f, nullptr, nullptr);
        if (s == 0) ln_residual<false><<<SEQ, 256, 0, stream>>>(Yo, X, gy, by, Ysum);
        else        ln_residual<true ><<<SEQ, 256, 0, stream>>>(Yo, X, gy, by, Ysum);
    }

    // ---- LSTM layer 0: Gx = Ysum @ Wih0^T + bih0 + bhh0, then recurrence ----
    wmma_gemm_nt<EP_BIAS | EP_BIAS2><<<GB, 128, 0, stream>>>(
        Ysum, Wih0, BIG, SEQ, G4, DM, 1.f, bih0, bhh0);
    zero_kernel<<<8, 256, 0, stream>>>(STATE, (size_t)(2 * HD + 64));
    lstm_recurrence<<<LSTM_NBLK, 256, LSTM_LDS_BYTES, stream>>>(
        BIG, Whh0, h0, hbuf, bar, SEQ);

    // ---- LSTM layer 1 ----
    wmma_gemm_nt<EP_BIAS | EP_BIAS2><<<GB, 128, 0, stream>>>(
        h0, Wih1, BIG, SEQ, G4, HD, 1.f, bih1, bhh1);
    zero_kernel<<<8, 256, 0, stream>>>(STATE, (size_t)(2 * HD + 64));
    lstm_recurrence<<<LSTM_NBLK, 256, LSTM_LDS_BYTES, stream>>>(
        BIG, Whh1, h1, hbuf, bar, SEQ);

    // ---- head: relu(h1 @ Wka^T + bka) -> LN -> sigmoid(dot Wkd + bkd) ----
    wmma_gemm_nt<EP_BIAS | EP_RELU><<<GB, 128, 0, stream>>>(
        h1, Wka, Ob, SEQ, HD, HD, 1.f, bka, nullptr);
    head_kernel<<<SEQ, 256, 0, stream>>>(Ob, gka, bka2, Wkd, bkd, out_o);
}
